// transformer_encoder_cell_1099511627871
// MI455X (gfx1250) — compile-verified, matrix operand order checked by reading, not run on hardware
//
#include <hip/hip_runtime.h>
#include <hip/hip_bf16.h>

// ---------------------------------------------------------------------------
// Transformer encoder cell for MI455X (gfx1250, wave32, WMMA).
// All GEMMs use v_wmma_f32_16x16x32_f16 (f16 inputs, f32 accumulation).
// ---------------------------------------------------------------------------

#define D_MODEL 768
#define D_FC    3072
#define NHEAD   12
#define HDIM    64
#define NB      4
#define SEQ     1024
#define LN_EPS  1e-5f

typedef __attribute__((ext_vector_type(16))) _Float16 v16h;
typedef __attribute__((ext_vector_type(8)))  float    v8f;

// --- WMMA helpers -----------------------------------------------------------

__device__ inline v8f wmma_f16(v16h a, v16h b, v8f c) {
  // (neg_a, A, neg_b, B, c_mod, C, reuse_a, reuse_b)
  return __builtin_amdgcn_wmma_f32_16x16x32_f16(false, a, false, b,
                                                (short)0, c, false, false);
}

// Load one 16x32 f16 fragment (row-major).  Lane l (0..15) holds row l; the
// lane half selects which 8-wide K slice goes to VGPRs 0..3 vs the +16 slice
// to VGPRs 4..7, per the CDNA5 16-bit A layout.  Caller bakes
// (row * ld + 8*half) into `p`; here we just do two b128 loads.
__device__ inline v16h load_frag16(const _Float16* p) {
  union { v16h v; float4 q[2]; } u;
  u.q[0] = *reinterpret_cast<const float4*>(p);       // K = 8h .. 8h+7
  u.q[1] = *reinterpret_cast<const float4*>(p + 16);  // K = 16+8h .. +7
  return u.v;
}

// --- Generic batched GEMM:  C[z] = A[z] (MxK) * Bt[z]^T (Bt is [N,K]) -------
// Wave tile (MI*16) x 64, 8 waves per block along M -> block tile MI*128 x 64.
// MI=4: 16 WMMAs per 8 fragment loads (feed-bandwidth optimal for big GEMMs).
// MI=2: smaller tile for the low-parallelism attention*V GEMM.
// flags: bit0 = store f16, bit1 = relu. bias (len N, f32) optional.

template <int MI>
__global__ __launch_bounds__(256) void gemm_wmma_f16k(
    const _Float16* __restrict__ A,  long long sA, int lda,
    const _Float16* __restrict__ Bt, long long sB, int ldb,
    void* __restrict__ Cv,           long long sC, int ldc,
    int M, int N, int K,
    const float* __restrict__ bias, int flags)
{
  const int lane = threadIdx.x & 31;
  const int wave = threadIdx.x >> 5;
  const int z    = blockIdx.z;

  A  += (long long)z * sA;
  Bt += (long long)z * sB;

  const int m0 = blockIdx.y * (MI * 16 * 8) + wave * (MI * 16);
  const int n0 = blockIdx.x * 64;
  if (m0 >= M || n0 >= N) return;

  const int l  = lane & 15;
  const int hh = lane >> 4;

  const _Float16* arow[MI];
#pragma unroll
  for (int mi = 0; mi < MI; ++mi)
    arow[mi] = A + (long long)(m0 + 16 * mi + l) * lda + 8 * hh;
  const _Float16* brow[4];
#pragma unroll
  for (int ni = 0; ni < 4; ++ni)
    brow[ni] = Bt + (long long)(n0 + 16 * ni + l) * ldb + 8 * hh;

  const v8f z8 = {0.f, 0.f, 0.f, 0.f, 0.f, 0.f, 0.f, 0.f};
  v8f acc[MI][4];
#pragma unroll
  for (int mi = 0; mi < MI; ++mi)
#pragma unroll
    for (int ni = 0; ni < 4; ++ni) acc[mi][ni] = z8;

  for (int k0 = 0; k0 < K; k0 += 32) {
    v16h a[MI];
#pragma unroll
    for (int mi = 0; mi < MI; ++mi) a[mi] = load_frag16(arow[mi] + k0);
#pragma unroll
    for (int ni = 0; ni < 4; ++ni) {
      const v16h bf = load_frag16(brow[ni] + k0);   // transient B fragment
#pragma unroll
      for (int mi = 0; mi < MI; ++mi)
        acc[mi][ni] = wmma_f16(a[mi], bf, acc[mi][ni]);
    }
  }

  // C/D layout: VGPR r, lane<16 -> (M=r, N=lane); lane>=16 -> (M=8+r, N=lane-16)
  float*    Cf = (float*)Cv    + (long long)z * sC;
  _Float16* Ch = (_Float16*)Cv + (long long)z * sC;
  const int col0 = lane & 15;
  const int rb   = (lane >> 4) * 8;
#pragma unroll
  for (int mi = 0; mi < MI; ++mi) {
#pragma unroll
    for (int ni = 0; ni < 4; ++ni) {
      const int col  = n0 + ni * 16 + col0;
      const float bv = bias ? bias[col] : 0.f;
#pragma unroll
      for (int r = 0; r < 8; ++r) {
        const long long row = m0 + mi * 16 + rb + r;
        float v = acc[mi][ni][r] + bv;
        if (flags & 2) v = fmaxf(v, 0.f);
        if (flags & 1) Ch[row * ldc + col] = (_Float16)v;
        else           Cf[row * ldc + col] = v;
      }
    }
  }
}

// --- Masked softmax: scores[H,S,S] f32 -> alpha[H,S,S] f16, 1 wave per row --

__global__ __launch_bounds__(256) void softmax_mask_k(
    const float* __restrict__ scores, const int* __restrict__ mask,
    _Float16* __restrict__ alpha)
{
  const int lane = threadIdx.x & 31;
  const int wave = threadIdx.x >> 5;
  const int row  = blockIdx.x * 8 + wave;       // [0, NHEAD*SEQ)
  const int q    = row & (SEQ - 1);
  const float scale = 0.28867513459481287f;     // 1/sqrt(H=12)

  const float* srow = scores + (long long)row * SEQ;
  const int*   mrow = mask   + (long long)q   * SEQ;

  float vals[32];
  float mx = -3.0e38f;
#pragma unroll
  for (int i = 0; i < 32; ++i) {
    const int k = lane + 32 * i;
    float v = srow[k] * scale;
    if (mrow[k] == 0) v = -1.0e9f;
    vals[i] = v;
    mx = fmaxf(mx, v);
  }
#pragma unroll
  for (int off = 16; off > 0; off >>= 1) mx = fmaxf(mx, __shfl_xor(mx, off, 32));

  float sum = 0.f;
#pragma unroll
  for (int i = 0; i < 32; ++i) {
    const float p = __expf(vals[i] - mx);
    vals[i] = p;
    sum += p;
  }
#pragma unroll
  for (int off = 16; off > 0; off >>= 1) sum += __shfl_xor(sum, off, 32);

  const float inv = 1.0f / sum;
  _Float16* arow = alpha + (long long)row * SEQ;
#pragma unroll
  for (int i = 0; i < 32; ++i) arow[lane + 32 * i] = (_Float16)(vals[i] * inv);
}

// --- Residual add + LayerNorm over 768-wide rows, 1 wave per row ------------

__global__ __launch_bounds__(256) void add_ln_k(
    const float* __restrict__ a, const float* __restrict__ res,
    const float* __restrict__ g, const float* __restrict__ be,
    float* __restrict__ outf, _Float16* __restrict__ outh)
{
  const int lane = threadIdx.x & 31;
  const int wave = threadIdx.x >> 5;
  const long long row = (long long)blockIdx.x * 8 + wave;
  const float* ar = a   + row * D_MODEL;
  const float* rr = res + row * D_MODEL;

  float vals[24];
  float s = 0.f;
#pragma unroll
  for (int i = 0; i < 24; ++i) {
    const int c = lane + 32 * i;
    const float v = ar[c] + rr[c];
    vals[i] = v;
    s += v;
  }
#pragma unroll
  for (int off = 16; off > 0; off >>= 1) s += __shfl_xor(s, off, 32);
  const float mu = s * (1.0f / D_MODEL);

  float vs = 0.f;
#pragma unroll
  for (int i = 0; i < 24; ++i) {
    const float d = vals[i] - mu;
    vs += d * d;
  }
#pragma unroll
  for (int off = 16; off > 0; off >>= 1) vs += __shfl_xor(vs, off, 32);
  const float rstd = rsqrtf(vs * (1.0f / D_MODEL) + LN_EPS);

#pragma unroll
  for (int i = 0; i < 24; ++i) {
    const int c = lane + 32 * i;
    const float y = (vals[i] - mu) * rstd * g[c] + be[c];
    if (outf) outf[row * D_MODEL + c] = y;
    if (outh) outh[row * D_MODEL + c] = (_Float16)y;
  }
}

// --- Elementwise convert / transpose helpers --------------------------------

__global__ void cvt_f32_f16_k(const float* __restrict__ in,
                              _Float16* __restrict__ out, long long n) {
  const long long i = (long long)blockIdx.x * 256 + threadIdx.x;
  if (i < n) out[i] = (_Float16)in[i];
}

// W[K,N] f32  ->  Wt[N,K] f16
__global__ void transpose_cvt_k(const float* __restrict__ W,
                                _Float16* __restrict__ Wt, int K, int N) {
  const long long i = (long long)blockIdx.x * 256 + threadIdx.x;
  if (i >= (long long)K * N) return;
  const int n = (int)(i / K);
  const int k = (int)(i % K);
  Wt[i] = (_Float16)W[(long long)k * N + n];
}

// Vh[B*S, 768] f16 -> Vt[B,H,64,S] f16
__global__ void transpose_v_k(const _Float16* __restrict__ Vh,
                              _Float16* __restrict__ Vt) {
  const long long i = (long long)blockIdx.x * 256 + threadIdx.x;
  if (i >= (long long)NB * SEQ * D_MODEL) return;
  const int c = (int)(i % D_MODEL);
  const long long bs = i / D_MODEL;
  const int s = (int)(bs % SEQ);
  const int b = (int)(bs / SEQ);
  const int h = c >> 6;
  const int d = c & 63;
  Vt[((long long)(b * NHEAD + h) * HDIM + d) * SEQ + s] = Vh[i];
}

// ---------------------------------------------------------------------------

extern "C" void kernel_launch(void* const* d_in, const int* in_sizes, int n_in,
                              void* d_out, int out_size, void* d_ws, size_t ws_size,
                              hipStream_t stream) {
  const float* x    = (const float*)d_in[0];
  const int*   mask = (const int*)  d_in[1];
  const float* WQ   = (const float*)d_in[2];
  const float* WK   = (const float*)d_in[3];
  const float* WV   = (const float*)d_in[4];
  const float* Wfc  = (const float*)d_in[5];
  const float* W1   = (const float*)d_in[6];
  const float* b1   = (const float*)d_in[7];
  const float* W2   = (const float*)d_in[8];
  const float* b2   = (const float*)d_in[9];
  const float* g1   = (const float*)d_in[10];
  const float* be1  = (const float*)d_in[11];
  const float* g2   = (const float*)d_in[12];
  const float* be2  = (const float*)d_in[13];
  float* out = (float*)d_out;

  // ---- workspace layout (256B aligned), ~197 MB total ----
  size_t off = 0;
  char* base = (char*)d_ws;
  auto alloc = [&](size_t elems, size_t esz) -> void* {
    void* r = base + off;
    off += (elems * esz + 255) & ~(size_t)255;
    return r;
  };
  const long long NX  = (long long)NB * SEQ * D_MODEL;   // 3,145,728
  _Float16* xh    = (_Float16*)alloc(NX, 2);
  _Float16* WQT   = (_Float16*)alloc((size_t)D_MODEL * D_MODEL, 2);
  _Float16* WKT   = (_Float16*)alloc((size_t)D_MODEL * D_MODEL, 2);
  _Float16* WVT   = (_Float16*)alloc((size_t)D_MODEL * D_MODEL, 2);
  _Float16* WfcT  = (_Float16*)alloc((size_t)D_MODEL * D_MODEL, 2);
  _Float16* W1T   = (_Float16*)alloc((size_t)D_FC * D_MODEL, 2);
  _Float16* W2T   = (_Float16*)alloc((size_t)D_MODEL * D_FC, 2);
  _Float16* Qh    = (_Float16*)alloc(NX, 2);
  _Float16* Kh    = (_Float16*)alloc(NX, 2);
  _Float16* Vh    = (_Float16*)alloc(NX, 2);
  _Float16* Vt    = (_Float16*)alloc(NX, 2);             // [B,H,64,S]
  float*    scores= (float*)   alloc((size_t)NHEAD * SEQ * SEQ, 4);  // per-b
  _Float16* alphah= (_Float16*)alloc((size_t)NHEAD * SEQ * SEQ, 2);  // per-b
  _Float16* atth  = (_Float16*)alloc(NX, 2);
  float*    attfc = (float*)   alloc(NX, 4);
  float*    normf = (float*)   alloc(NX, 4);
  _Float16* normh = (_Float16*)alloc(NX, 2);
  _Float16* h1    = (_Float16*)alloc((size_t)NB * SEQ * D_FC, 2);
  float*    lin   = (float*)   alloc(NX, 4);

  const dim3 blk(256);

  // ---- 1) convert x, transpose+convert weights ----
  cvt_f32_f16_k<<<(unsigned)((NX + 255) / 256), blk, 0, stream>>>(x, xh, NX);
  const long long nw = (long long)D_MODEL * D_MODEL;
  transpose_cvt_k<<<(unsigned)((nw + 255) / 256), blk, 0, stream>>>(WQ,  WQT,  D_MODEL, D_MODEL);
  transpose_cvt_k<<<(unsigned)((nw + 255) / 256), blk, 0, stream>>>(WK,  WKT,  D_MODEL, D_MODEL);
  transpose_cvt_k<<<(unsigned)((nw + 255) / 256), blk, 0, stream>>>(WV,  WVT,  D_MODEL, D_MODEL);
  transpose_cvt_k<<<(unsigned)((nw + 255) / 256), blk, 0, stream>>>(Wfc, WfcT, D_MODEL, D_MODEL);
  const long long nw1 = (long long)D_MODEL * D_FC;
  transpose_cvt_k<<<(unsigned)((nw1 + 255) / 256), blk, 0, stream>>>(W1, W1T, D_MODEL, D_FC);
  transpose_cvt_k<<<(unsigned)((nw1 + 255) / 256), blk, 0, stream>>>(W2, W2T, D_FC, D_MODEL);

  // ---- 2) QKV projections: [4096,768] = [4096,768] x [768,768], f16 out ----
  const dim3 g_qkv(D_MODEL / 64, (NB * SEQ) / 512, 1);   // (12,8,1)
  gemm_wmma_f16k<4><<<g_qkv, blk, 0, stream>>>(xh, 0, D_MODEL, WQT, 0, D_MODEL,
                                               Qh, 0, D_MODEL, NB * SEQ, D_MODEL, D_MODEL,
                                               nullptr, 1);
  gemm_wmma_f16k<4><<<g_qkv, blk, 0, stream>>>(xh, 0, D_MODEL, WKT, 0, D_MODEL,
                                               Kh, 0, D_MODEL, NB * SEQ, D_MODEL, D_MODEL,
                                               nullptr, 1);
  gemm_wmma_f16k<4><<<g_qkv, blk, 0, stream>>>(xh, 0, D_MODEL, WVT, 0, D_MODEL,
                                               Vh, 0, D_MODEL, NB * SEQ, D_MODEL, D_MODEL,
                                               nullptr, 1);
  transpose_v_k<<<(unsigned)((NX + 255) / 256), blk, 0, stream>>>(Vh, Vt);

  // ---- 3) attention, per batch (scores/alpha buffers reused, stream-ordered)
  for (int b = 0; b < NB; ++b) {
    const _Float16* Qb = Qh + (long long)b * SEQ * D_MODEL;
    const _Float16* Kb = Kh + (long long)b * SEQ * D_MODEL;
    // scores[h] = Q_bh (S x 64) * K_bh^T ; K_bh rows serve as [N,K] operand
    const dim3 g_sc(SEQ / 64, SEQ / 512, NHEAD);         // (16,2,12)
    gemm_wmma_f16k<4><<<g_sc, blk, 0, stream>>>(
        Qb, HDIM, D_MODEL, Kb, HDIM, D_MODEL,
        scores, (long long)SEQ * SEQ, SEQ,
        SEQ, SEQ, HDIM, nullptr, 0);
    softmax_mask_k<<<(NHEAD * SEQ) / 8, blk, 0, stream>>>(
        scores, mask + (long long)b * SEQ * SEQ, alphah);
    // att[h] = alpha_h (S x S) * V_bh ; Vt[b,h] is [64,S] = [N,K]
    const dim3 g_av(HDIM / 64, SEQ / 256, NHEAD);        // (1,4,12)
    gemm_wmma_f16k<2><<<g_av, blk, 0, stream>>>(
        alphah, (long long)SEQ * SEQ, SEQ,
        Vt + (long long)b * NHEAD * HDIM * SEQ, (long long)HDIM * SEQ, SEQ,
        atth + (long long)b * SEQ * D_MODEL, HDIM, D_MODEL,
        SEQ, HDIM, SEQ, nullptr, 1);
  }

  // ---- 4) att @ Wfc (f32 out), residual + LN1 ----
  gemm_wmma_f16k<4><<<g_qkv, blk, 0, stream>>>(atth, 0, D_MODEL, WfcT, 0, D_MODEL,
                                               attfc, 0, D_MODEL, NB * SEQ, D_MODEL, D_MODEL,
                                               nullptr, 0);
  add_ln_k<<<(NB * SEQ) / 8, blk, 0, stream>>>(attfc, x, g1, be1, normf, normh);

  // ---- 5) FFN: relu(norm @ W1 + b1) @ W2 + b2 ----
  const dim3 g_f1(D_FC / 64, (NB * SEQ) / 512, 1);       // (48,8,1)
  gemm_wmma_f16k<4><<<g_f1, blk, 0, stream>>>(normh, 0, D_MODEL, W1T, 0, D_MODEL,
                                              h1, 0, D_FC, NB * SEQ, D_FC, D_MODEL,
                                              b1, 1 | 2);
  const dim3 g_f2(D_MODEL / 64, (NB * SEQ) / 512, 1);    // (12,8,1)
  gemm_wmma_f16k<4><<<g_f2, blk, 0, stream>>>(h1, 0, D_FC, W2T, 0, D_FC,
                                              lin, 0, D_MODEL, NB * SEQ, D_MODEL, D_FC,
                                              b2, 0);

  // ---- 6) residual + LN2 -> d_out ----
  add_ln_k<<<(NB * SEQ) / 8, blk, 0, stream>>>(lin, normf, g2, be2, out, nullptr);
  (void)in_sizes; (void)n_in; (void)out_size; (void)ws_size;
}
